// MultiHeadAttention_29377576304748
// MI455X (gfx1250) — matrix operand order repaired and checked by reading, not compiled, not run on hardware
//
#include <hip/hip_runtime.h>

#define NH   8
#define HD   64
#define DIN  512
#define DOUT 512
#define BB   4
#define SS   2048
#define LOG2E 1.44269504088896340736f

typedef __attribute__((ext_vector_type(16))) __bf16 v16bf;
typedef __attribute__((ext_vector_type(8)))  float  v8f;
typedef __attribute__((ext_vector_type(4)))  unsigned int u32x4;
typedef __attribute__((ext_vector_type(4)))  int i32x4;
typedef __attribute__((ext_vector_type(8)))  int i32x8;

#if defined(__has_builtin)
#  if __has_builtin(__builtin_amdgcn_tensor_load_to_lds)
#    define USE_TDM 1
#  endif
#endif
#ifndef USE_TDM
#  define USE_TDM 0
#endif

struct B32 { uint4 a, b; };

__device__ __forceinline__ unsigned short f2bf(float f) {
  return __builtin_bit_cast(unsigned short, (__bf16)f);   // native v_cvt
}
__device__ __forceinline__ unsigned int pk2(float lo, float hi) {
  return (unsigned int)f2bf(lo) | ((unsigned int)f2bf(hi) << 16);
}
__device__ __forceinline__ float fexp2(float x) { return __builtin_amdgcn_exp2f(x); }

__device__ __forceinline__ v16bf ld16bf(const unsigned short* p) {
  B32 t; const uint4* q = (const uint4*)p; t.a = q[0]; t.b = q[1];
  return __builtin_bit_cast(v16bf, t);
}
__device__ __forceinline__ v16bf ld2x8bf(const unsigned short* p0, const unsigned short* p1) {
  B32 t; t.a = *(const uint4*)p0; t.b = *(const uint4*)p1;
  return __builtin_bit_cast(v16bf, t);
}

#if USE_TDM
// Build a 2D tensor-DMA descriptor (D#) per CDNA5 ISA 8.3/8.4 and issue TENSOR_LOAD_TO_LDS.
// dims/strides in data_size=2B elements; lds_addr is a byte offset into the WG's LDS.
__device__ __forceinline__ void tdm_load_2d(unsigned lds_addr, const void* gptr,
                                            unsigned tile_d0, unsigned tile_d1,
                                            unsigned ten_d0, unsigned ten_d1,
                                            unsigned long long stride0) {
  unsigned long long ga = (unsigned long long)(size_t)gptr;
  u32x4 g0;
  g0[0] = 1u;                                            // count=1, user descriptor
  g0[1] = lds_addr;                                      // lds_addr [63:32]
  g0[2] = (unsigned)(ga & 0xffffffffu);                  // global_addr [95:64]
  g0[3] = (unsigned)((ga >> 32) & 0x1ffffffu) | (2u << 30);  // addr[56:32] | type=2
  i32x8 g1;
  g1[0] = (int)(1u << 16);                               // data_size=1 (2 bytes)
  g1[1] = (int)(ten_d0 << 16);                           // tensor_dim0[15:0]
  g1[2] = (int)((ten_d0 >> 16) | (ten_d1 << 16));        // dim0[31:16] | dim1[15:0]
  g1[3] = (int)((ten_d1 >> 16) | (tile_d0 << 16));       // dim1[31:16] | tile_dim0
  g1[4] = (int)(tile_d1 & 0xffffu);                      // tile_dim1 (tile_dim2=0 -> 2D)
  g1[5] = (int)(unsigned)(stride0 & 0xffffffffu);        // tensor_dim0_stride[31:0]
  g1[6] = (int)(unsigned)((stride0 >> 32) & 0xffffu);    // stride[47:32] (dim1_stride=0)
  g1[7] = 0;
  i32x4 z4 = {0, 0, 0, 0};
#if __clang_major__ >= 23
  i32x8 z8 = {0, 0, 0, 0, 0, 0, 0, 0};
  __builtin_amdgcn_tensor_load_to_lds(g0, g1, z4, z4, z8, 0);
#else
  __builtin_amdgcn_tensor_load_to_lds(g0, g1, z4, z4, 0);
#endif
}
__device__ __forceinline__ unsigned ldsoff(const void* p) {
  return (unsigned)(size_t)p;   // generic->LDS: low 32 bits are the addrspace(3) offset
}
#endif

// W[K][N] f32 -> Wt[N][K] bf16
__global__ void wtrans_kernel(const float* __restrict__ w, unsigned short* __restrict__ wt) {
  int idx = blockIdx.x * 256 + threadIdx.x;
  int k = idx >> 9, n = idx & 511;
  wt[(size_t)n * DIN + k] = f2bf(w[idx]);
}

// X f32 -> bf16, once per matrix
__global__ void xcvt_kernel(const float* __restrict__ x, unsigned short* __restrict__ xb) {
  size_t i = ((size_t)blockIdx.x * 256 + threadIdx.x) * 8;
  float4 f0 = ((const float4*)(x + i))[0];
  float4 f1 = ((const float4*)(x + i))[1];
  uint4 o = make_uint4(pk2(f0.x, f0.y), pk2(f0.z, f0.w), pk2(f1.x, f1.y), pk2(f1.z, f1.w));
  *(uint4*)(xb + i) = o;
}

// One wave: 16 rows x 64 cols (one head) of C = Xb @ W.  The 4 waves of a block share
// head h, so the head's 64x512 Wt panel (64KB) is TDM-staged to LDS once and shared.
__global__ __launch_bounds__(128)
void proj_kernel(const unsigned short* __restrict__ Xb, const unsigned short* __restrict__ Wt,
                 unsigned short* __restrict__ out, float scale, int vtrans) {
#if USE_TDM
  __shared__ __align__(16) unsigned short Wlds[64][512];
#endif
  const int w    = threadIdx.x >> 5;
  const int lane = threadIdx.x & 31;
  const int lr   = lane & 15;
  const int half = lane >> 4;
  const int row0 = (blockIdx.x * 4 + w) * 16;
  const int h    = blockIdx.y;

#if USE_TDM
  if (w == 0)
    tdm_load_2d(ldsoff(&Wlds[0][0]), Wt + (size_t)h * HD * DIN, DIN, 64, DIN, 64, DIN);
#endif

  v8f acc[4];
  #pragma unroll
  for (int t = 0; t < 4; ++t)
    #pragma unroll
    for (int i = 0; i < 8; ++i) acc[t][i] = 0.f;

  const unsigned short* xrow = Xb + (size_t)(row0 + lr) * DIN;

#if USE_TDM
  if (w == 0) __builtin_amdgcn_s_wait_tensorcnt(0);
  __syncthreads();
#endif

  for (int kk = 0; kk < DIN; kk += 32) {
    v16bf a = ld2x8bf(xrow + kk + half * 16, xrow + kk + half * 16 + 8);
    #pragma unroll
    for (int t = 0; t < 4; ++t) {
#if USE_TDM
      const unsigned short* bp = &Wlds[t * 16 + lr][kk + half * 16];
#else
      const unsigned short* bp = Wt + (size_t)(h * HD + t * 16 + lr) * DIN + kk + half * 16;
#endif
      v16bf b = ld16bf(bp);
      acc[t] = __builtin_amdgcn_wmma_f32_16x16x32_bf16(false, a, false, b,
                                                       (short)0, acc[t], false, false);
    }
  }

  const int bidx = row0 / SS;
  const int s0   = (row0 % SS) + half * 8;
  if (vtrans == 0) {
    #pragma unroll
    for (int t = 0; t < 4; ++t) {
      const int d = t * 16 + lr;
      unsigned short* op = out + ((size_t)(bidx * NH + h) * SS + s0) * HD + d;
      #pragma unroll
      for (int v = 0; v < 8; ++v) op[(size_t)v * HD] = f2bf(acc[t][v] * scale);
    }
  } else {
    #pragma unroll
    for (int t = 0; t < 4; ++t) {
      const int d = t * 16 + lr;
      unsigned short* op = out + ((size_t)(bidx * NH + h) * HD + d) * SS + s0;
      uint4 p = make_uint4(pk2(acc[t][0] * scale, acc[t][1] * scale),
                           pk2(acc[t][2] * scale, acc[t][3] * scale),
                           pk2(acc[t][4] * scale, acc[t][5] * scale),
                           pk2(acc[t][6] * scale, acc[t][7] * scale));
      *(uint4*)op = p;
    }
  }
}

// Flash attention. 4 waves/block share one (b,h): K/V 64-key tiles staged to LDS by the
// Tensor Data Mover (wave 0, double-buffered, TENSORcnt-tracked), consumed by all waves.
__global__ __launch_bounds__(128)
void attn_kernel(const unsigned short* __restrict__ Q,
                 const unsigned short* __restrict__ K,
                 const unsigned short* __restrict__ Vt,
                 const float* __restrict__ mask,
                 float* __restrict__ out) {
#if USE_TDM
  __shared__ __align__(16) unsigned short Klds[2][64][64];   // [key][d]
  __shared__ __align__(16) unsigned short Vlds[2][64][64];   // [d][key]
#endif
  __shared__ __align__(16) unsigned short Pbuf[4][16][64];
  const int w    = threadIdx.x >> 5;
  const int lane = threadIdx.x & 31;
  const int lr   = lane & 15;
  const int half = lane >> 4;
  const int bh   = blockIdx.x;
  const int q0   = (blockIdx.y * 4 + w) * 16;
  const int nsteps = SS / 64;

  const unsigned short* Qh = Q  + (size_t)bh * SS * HD;
  const unsigned short* Kh = K  + (size_t)bh * SS * HD;
  const unsigned short* Vh = Vt + (size_t)bh * HD * SS;
  const float* mrow = mask + (size_t)(bh / NH) * SS;

#if USE_TDM
  if (w == 0) {   // prologue: fill both buffers (4 TDM ops in flight max)
    tdm_load_2d(ldsoff(&Klds[0][0][0]), Kh, 64, 64, 64, 64, 64);
    tdm_load_2d(ldsoff(&Vlds[0][0][0]), Vh, 64, 64, 64, 64, SS);
    tdm_load_2d(ldsoff(&Klds[1][0][0]), Kh + (size_t)64 * HD, 64, 64, 64, 64, 64);
    tdm_load_2d(ldsoff(&Vlds[1][0][0]), Vh + 64,              64, 64, 64, 64, SS);
  }
#endif

  const unsigned short* qrow = Qh + (size_t)(q0 + lr) * HD;
  v16bf aQ0 = ld2x8bf(qrow + half * 16,      qrow + 8 + half * 16);
  v16bf aQ1 = ld2x8bf(qrow + 32 + half * 16, qrow + 40 + half * 16);

  float mi[8], li[8];
  v8f o[4];
  #pragma unroll
  for (int v = 0; v < 8; ++v) { mi[v] = -3.0e38f; li[v] = 0.f; }
  #pragma unroll
  for (int t = 0; t < 4; ++t)
    #pragma unroll
    for (int i = 0; i < 8; ++i) o[t][i] = 0.f;

  unsigned short (*pl)[64] = Pbuf[w];

  for (int i = 0; i < nsteps; ++i) {
    const int k0 = i * 64;
#if USE_TDM
    const int buf = i & 1;
    if (w == 0) {
      if (i < nsteps - 1) __builtin_amdgcn_s_wait_tensorcnt(2);  // oldest K+V pair done
      else                __builtin_amdgcn_s_wait_tensorcnt(0);
    }
    __syncthreads();
#endif
    // ---- logits (log2 domain; mask folded into WMMA C operand) ----
    v8f s[4];
    #pragma unroll
    for (int t = 0; t < 4; ++t) {
#if USE_TDM
      const unsigned short* krow = &Klds[buf][t * 16 + lr][0];
#else
      const unsigned short* krow = Kh + (size_t)(k0 + t * 16 + lr) * HD;
#endif
      v16bf b0 = ld16bf(krow + half * 16);
      v16bf b1 = ld16bf(krow + 32 + half * 16);
      float mv = mrow[k0 + t * 16 + lr] * LOG2E;
      v8f z;
      #pragma unroll
      for (int j = 0; j < 8; ++j) z[j] = mv;
      z = __builtin_amdgcn_wmma_f32_16x16x32_bf16(false, aQ0, false, b0, (short)0, z, false, false);
      z = __builtin_amdgcn_wmma_f32_16x16x32_bf16(false, aQ1, false, b1, (short)0, z, false, false);
      s[t] = z;
    }
    // ---- online softmax ----
    float alpha[8], rs[8];
    #pragma unroll
    for (int v = 0; v < 8; ++v) {
      float x = fmaxf(fmaxf(s[0][v], s[1][v]), fmaxf(s[2][v], s[3][v]));
      x = fmaxf(x, __shfl_xor(x, 1, 32));
      x = fmaxf(x, __shfl_xor(x, 2, 32));
      x = fmaxf(x, __shfl_xor(x, 4, 32));
      x = fmaxf(x, __shfl_xor(x, 8, 32));
      float mn = fmaxf(mi[v], x);
      alpha[v] = fexp2(mi[v] - mn);
      mi[v] = mn;
      rs[v] = 0.f;
    }
    #pragma unroll
    for (int t = 0; t < 4; ++t)
      #pragma unroll
      for (int v = 0; v < 8; ++v) {
        float p = fexp2(s[t][v] - mi[v]);
        rs[v] += p;
        pl[half * 8 + v][t * 16 + lr] = f2bf(p);
      }
    #pragma unroll
    for (int v = 0; v < 8; ++v) {
      float r = rs[v];
      r += __shfl_xor(r, 1, 32);
      r += __shfl_xor(r, 2, 32);
      r += __shfl_xor(r, 4, 32);
      r += __shfl_xor(r, 8, 32);
      li[v] = li[v] * alpha[v] + r;
    }
    #pragma unroll
    for (int t = 0; t < 4; ++t)
      #pragma unroll
      for (int v = 0; v < 8; ++v) o[t][v] *= alpha[v];

    asm volatile("s_wait_dscnt 0" ::: "memory");   // wave-synchronous P staging
    v16bf aP0 = ld2x8bf(&pl[lr][half * 16],      &pl[lr][8 + half * 16]);
    v16bf aP1 = ld2x8bf(&pl[lr][32 + half * 16], &pl[lr][40 + half * 16]);

    // ---- O += P @ V ----
    #pragma unroll
    for (int t = 0; t < 4; ++t) {
#if USE_TDM
      const unsigned short* vrow = &Vlds[buf][t * 16 + lr][0];
#else
      const unsigned short* vrow = Vh + (size_t)(t * 16 + lr) * SS + k0;
#endif
      v16bf bv0 = ld16bf(vrow + half * 16);
      v16bf bv1 = ld16bf(vrow + 32 + half * 16);
      o[t] = __builtin_amdgcn_wmma_f32_16x16x32_bf16(false, aP0, false, bv0, (short)0, o[t], false, false);
      o[t] = __builtin_amdgcn_wmma_f32_16x16x32_bf16(false, aP1, false, bv1, (short)0, o[t], false, false);
    }
#if USE_TDM
    __syncthreads();                                // all waves done with buf
    if (w == 0 && i + 2 < nsteps) {                 // refill it for step i+2
      tdm_load_2d(ldsoff(&Klds[buf][0][0]), Kh + (size_t)(k0 + 128) * HD, 64, 64, 64, 64, 64);
      tdm_load_2d(ldsoff(&Vlds[buf][0][0]), Vh + (k0 + 128),              64, 64, 64, 64, SS);
    }
#endif
  }

  const int bidx = bh / NH, hh = bh % NH;
  #pragma unroll
  for (int v = 0; v < 8; ++v) {
    float inv = 1.f / li[v];
    const int srow = q0 + half * 8 + v;
    float* orow = out + ((size_t)bidx * SS + srow) * DOUT + hh * HD;
    #pragma unroll
    for (int t = 0; t < 4; ++t) orow[t * 16 + lr] = o[t][v] * inv;
  }
}

extern "C" void kernel_launch(void* const* d_in, const int* in_sizes, int n_in,
                              void* d_out, int out_size, void* d_ws, size_t ws_size,
                              hipStream_t stream) {
  const float* key   = (const float*)d_in[0];
  const float* mask  = (const float*)d_in[1];
  const float* query = (const float*)d_in[2];
  const float* value = (const float*)d_in[3];
  const float* wq    = (const float*)d_in[4];
  const float* wk    = (const float*)d_in[5];
  const float* wv    = (const float*)d_in[6];
  float* out = (float*)d_out;

  unsigned short* ws    = (unsigned short*)d_ws;
  unsigned short* wtbuf = ws;
  unsigned short* xbuf  = wtbuf + (size_t)DIN * DOUT;
  unsigned short* qb    = xbuf + (size_t)BB * SS * DIN;
  unsigned short* kb    = qb + (size_t)BB * NH * SS * HD;
  unsigned short* vtb   = kb + (size_t)BB * NH * SS * HD;

  const float qscale = 0.125f * LOG2E;
  const int xblocks = (BB * SS * DIN) / (256 * 8);
  dim3 pg((BB * SS) / 64, NH);

  wtrans_kernel<<<(DIN * DOUT) / 256, 256, 0, stream>>>(wq, wtbuf);
  xcvt_kernel<<<xblocks, 256, 0, stream>>>(query, xbuf);
  proj_kernel<<<pg, 128, 0, stream>>>(xbuf, wtbuf, qb, qscale, 0);

  wtrans_kernel<<<(DIN * DOUT) / 256, 256, 0, stream>>>(wk, wtbuf);
  xcvt_kernel<<<xblocks, 256, 0, stream>>>(key, xbuf);
  proj_kernel<<<pg, 128, 0, stream>>>(xbuf, wtbuf, kb, 1.0f, 0);

  wtrans_kernel<<<(DIN * DOUT) / 256, 256, 0, stream>>>(wv, wtbuf);
  xcvt_kernel<<<xblocks, 256, 0, stream>>>(value, xbuf);
  proj_kernel<<<pg, 128, 0, stream>>>(xbuf, wtbuf, vtb, 1.0f, 1);

  dim3 ag(BB * NH, SS / 64);
  attn_kernel<<<ag, 128, 0, stream>>>(qb, kb, vtb, mask, out);
}
